// SpatioTemporalMLPAttention_14130442404183
// MI455X (gfx1250) — compile-verified, matrix operand order checked by reading, not run on hardware
//
#include <hip/hip_runtime.h>
#include <hip/hip_bf16.h>
#include <math.h>

// Problem constants (match reference)
#define B_   4
#define Q_   512
#define K_   512
#define F_   32
#define DV_  32
#define H_   64
#define H1_  256
#define H2_  64

typedef __attribute__((ext_vector_type(16))) _Float16 v16h;
typedef __attribute__((ext_vector_type(8)))  float    v8f;

__device__ __forceinline__ float gelu_f(float x) {
    // jax.nn.gelu (approximate=True, tanh form)
    const float c = 0.7978845608028654f;
    float x3 = x * x * x;
    return 0.5f * x * (1.0f + tanhf(c * (x + 0.044715f * x3)));
}

// Load one 16-half WMMA fragment whose lane data is two contiguous 8-half runs:
// halves 0-7 at p[0..7], halves 8-15 at p[16..23]  ->  two ds_load_b128.
__device__ __forceinline__ v16h load_frag(const _Float16* __restrict__ p) {
    v16h f;
    #pragma unroll
    for (int i = 0; i < 8; ++i) { f[i] = p[i]; f[8 + i] = p[16 + i]; }
    return f;
}

// a_q[b,q,c] = qs[b,q,:]@Ws1[0:32,c] + vnode[b,:]@Ws1[64:96,c] + bs1[c]
__global__ __launch_bounds__(256)
void precompute_aq_kernel(const float* __restrict__ qs, const float* __restrict__ vnode,
                          const float* __restrict__ Ws1, const float* __restrict__ bs1,
                          float* __restrict__ a_q) {
    int gid = blockIdx.x * 256 + threadIdx.x;   // over B*Q*H1
    int c   = gid & 255;
    int bq  = gid >> 8;
    int b   = bq / Q_;
    float acc = bs1[c];
    const float* qrow = qs + (size_t)bq * F_;
    const float* vrow = vnode + (size_t)b * DV_;
    #pragma unroll
    for (int f = 0; f < F_; ++f)  acc += qrow[f] * Ws1[f * H1_ + c];
    #pragma unroll
    for (int f = 0; f < DV_; ++f) acc += vrow[f] * Ws1[(64 + f) * H1_ + c];
    a_q[gid] = acc;
}

// a_k[b,k,c] = ks[b,k,:]@Ws1[32:64,c]
__global__ __launch_bounds__(256)
void precompute_ak_kernel(const float* __restrict__ ks, const float* __restrict__ Ws1,
                          float* __restrict__ a_k) {
    int gid = blockIdx.x * 256 + threadIdx.x;   // over B*K*H1
    int c   = gid & 255;
    int bk  = gid >> 8;
    float acc = 0.0f;
    const float* krow = ks + (size_t)bk * F_;
    #pragma unroll
    for (int f = 0; f < F_; ++f) acc += krow[f] * Ws1[(32 + f) * H1_ + c];
    a_k[gid] = acc;
}

// vproj[b,k,h] = vs[b,k,:]@Wv[:,h] + bv[h]
__global__ __launch_bounds__(256)
void precompute_vproj_kernel(const float* __restrict__ vs, const float* __restrict__ Wv,
                             const float* __restrict__ bv, float* __restrict__ vproj) {
    int gid = blockIdx.x * 256 + threadIdx.x;   // over B*K*H
    int h   = gid & 63;
    int bk  = gid >> 6;
    float acc = bv[h];
    const float* vrow = vs + (size_t)bk * F_;
    #pragma unroll
    for (int f = 0; f < F_; ++f) acc += vrow[f] * Wv[f * H_ + h];
    vproj[gid] = acc;
}

// Main fused kernel: per (b, 16q-tile, 16k-tile) block computes masked pre-softmax
// scores using WMMA for the H1->H2 layer (M=256 pairs, K=256, N=64).
__global__ __launch_bounds__(256)
void stmlp_scores_kernel(const float* __restrict__ a_q,
                         const float* __restrict__ a_k,
                         const float* __restrict__ Ws1,
                         const float* __restrict__ Ws2,
                         const float* __restrict__ bs2,
                         const float* __restrict__ Ws3,
                         const float* __restrict__ bs3,
                         const float* __restrict__ qs_s,
                         const float* __restrict__ ks_s,
                         const float* __restrict__ qs_t,
                         const float* __restrict__ ks_t,
                         const int*   __restrict__ valid_lens,
                         float* __restrict__ scores) {
    __shared__ _Float16 sH1[256][64];   // 256 pairs x 64 H1-cols (current chunk), fp16
    __shared__ _Float16 sW2T[64][64];   // Ws2 chunk TRANSPOSED: [H2-col][H1-row], fp16
    __shared__ float    sW1x[5][256];   // Ws1 rows 96..100 (geometric features)
    __shared__ float    sW3[64];
    __shared__ float    sB2[64];
    __shared__ float    sScore[256];

    const int t    = threadIdx.x;
    const int lane = t & 31;
    const int wv   = t >> 5;            // 8 waves
    const int kTile = blockIdx.x, qTile = blockIdx.y, b = blockIdx.z;
    const int qg = qTile * 16 + (t >> 4);
    const int kg = kTile * 16 + (t & 15);

    // Stage per-block constants
    for (int i = t; i < 5 * 256; i += 256)
        sW1x[i >> 8][i & 255] = Ws1[(96 + (i >> 8)) * H1_ + (i & 255)];
    if (t < 64) { sW3[t] = Ws3[t]; sB2[t] = bs2[t]; }
    sScore[t] = 0.0f;

    // Per-pair geometric features (thread t <-> pair t of this 16x16 tile)
    float sdx   = qs_s[(b * Q_ + qg) * 2 + 0] - ks_s[(b * K_ + kg) * 2 + 0];
    float sdy   = qs_s[(b * Q_ + qg) * 2 + 1] - ks_s[(b * K_ + kg) * 2 + 1];
    float sdsq  = sdx * sdx + sdy * sdy;
    float sdist = sqrtf(sdsq);
    float tdiff = qs_t[b * Q_ + qg] - ks_t[b * K_ + kg];

    const float* aq = a_q + (size_t)(b * Q_ + qg) * H1_;
    const float* ak = a_k + (size_t)(b * K_ + kg) * H1_;

    v8f acc[8];   // acc[pt*4+nt]: pairTile = wv*2+pt, nTile = nt
    {
        v8f z = {0.f, 0.f, 0.f, 0.f, 0.f, 0.f, 0.f, 0.f};
        #pragma unroll
        for (int r = 0; r < 8; ++r) acc[r] = z;
    }

    const int rowSel = (lane >> 4) & 1;     // hi half-wave handles K base +8
    const int ln16   = lane & 15;

    // Loop over H1 in chunks of 64
    for (int ch = 0; ch < 4; ++ch) {
        __syncthreads();   // previous chunk's WMMA reads done before restaging

        // Stage Ws2 chunk TRANSPOSED as fp16: sW2T[n][c] = Ws2[ch*64+c][n]
        for (int i = t; i < 64 * 64; i += 256) {
            int n = i >> 6, c = i & 63;
            sW2T[n][c] = (_Float16)Ws2[(ch * 64 + c) * H2_ + n];
        }
        // Layer-1: h1 = gelu(a_q + a_k + geom@Ws1[96:101]) for this pair, 64 cols
        #pragma unroll 8
        for (int c = 0; c < 64; ++c) {
            int col = ch * 64 + c;
            float v = aq[col] + ak[col]
                    + sdx   * sW1x[0][col] + sdy  * sW1x[1][col]
                    + sdist * sW1x[2][col] + sdsq * sW1x[3][col]
                    + tdiff * sW1x[4][col];
            sH1[t][c] = (_Float16)gelu_f(v);
        }
        __syncthreads();

        // Hoisted fragment loads: 4 distinct A frags, 8 distinct B frags per wave
        v16h afr[2][2];   // [pairTile-local][kk]
        v16h bfr[4][2];   // [nTile][kk]
        #pragma unroll
        for (int pt = 0; pt < 2; ++pt)
            #pragma unroll
            for (int kk = 0; kk < 2; ++kk)
                afr[pt][kk] = load_frag(&sH1[(wv * 2 + pt) * 16 + ln16][kk * 32 + rowSel * 8]);
        #pragma unroll
        for (int nt = 0; nt < 4; ++nt)
            #pragma unroll
            for (int kk = 0; kk < 2; ++kk)
                bfr[nt][kk] = load_frag(&sW2T[nt * 16 + ln16][kk * 32 + rowSel * 8]);

        // 16 back-to-back WMMAs per wave per chunk
        #pragma unroll
        for (int pt = 0; pt < 2; ++pt)
            #pragma unroll
            for (int nt = 0; nt < 4; ++nt)
                #pragma unroll
                for (int kk = 0; kk < 2; ++kk)
                    acc[pt * 4 + nt] = __builtin_amdgcn_wmma_f32_16x16x32_f16(
                        false, afr[pt][kk], false, bfr[nt][kk],
                        (short)0, acc[pt * 4 + nt], false, false);
    }

    // Layer-3: score[pair] = sum_n gelu(h2[pair,n] + bs2[n]) * Ws3[n]
    const int rowOff = rowSel * 8;
    #pragma unroll
    for (int pt = 0; pt < 2; ++pt) {
        #pragma unroll
        for (int nt = 0; nt < 4; ++nt) {
            int col = nt * 16 + ln16;
            float w3 = sW3[col], b2 = sB2[col];
            #pragma unroll
            for (int j = 0; j < 8; ++j) {
                int p = (wv * 2 + pt) * 16 + j + rowOff;   // D layout: VGPR j -> row j (+8 hi lanes)
                atomicAdd(&sScore[p], gelu_f(acc[pt * 4 + nt][j] + b2) * w3);
            }
        }
    }
    __syncthreads();

    float sc = sScore[t] + bs3[0];
    bool valid = (kg < valid_lens[b]) && (tdiff >= 0.0f) && (sdist <= 1e30f);
    scores[(size_t)(b * Q_ + qg) * K_ + kg] = valid ? sc : -INFINITY;
}

// Fused softmax over K + ctx = attn @ vproj, one workgroup per (b,q)
__global__ __launch_bounds__(256)
void softmax_ctx_kernel(const float* __restrict__ scores,
                        const float* __restrict__ vproj,
                        float* __restrict__ attn,
                        float* __restrict__ ctx) {
    __shared__ float sA[K_];
    __shared__ float red[256];
    int bq = blockIdx.x;
    int b  = bq / Q_;
    int t  = threadIdx.x;
    const float* srow = scores + (size_t)bq * K_;

    float m = -INFINITY;
    for (int k = t; k < K_; k += 256) m = fmaxf(m, srow[k]);
    red[t] = m; __syncthreads();
    for (int off = 128; off > 0; off >>= 1) {
        if (t < off) red[t] = fmaxf(red[t], red[t + off]);
        __syncthreads();
    }
    m = red[0];
    float mg = (m == -INFINITY) ? 0.0f : m;
    __syncthreads();

    float sum = 0.0f;
    for (int k = t; k < K_; k += 256) {
        float s = srow[k];
        float e = (s == -INFINITY) ? 0.0f : __expf(s - mg);
        sA[k] = e; sum += e;
    }
    red[t] = sum; __syncthreads();
    for (int off = 128; off > 0; off >>= 1) {
        if (t < off) red[t] += red[t + off];
        __syncthreads();
    }
    float inv = (red[0] > 0.0f) ? (1.0f / red[0]) : 0.0f;
    __syncthreads();

    for (int k = t; k < K_; k += 256) {
        float a = sA[k] * inv;
        sA[k] = a;
        attn[(size_t)bq * K_ + k] = a;
    }
    __syncthreads();

    if (t < H_) {
        float accv = 0.0f;
        const float* vp = vproj + (size_t)b * K_ * H_ + t;
        #pragma unroll 8
        for (int k = 0; k < K_; ++k) accv += sA[k] * vp[k * H_];
        ctx[(size_t)bq * H_ + t] = accv;
    }
}

extern "C" void kernel_launch(void* const* d_in, const int* in_sizes, int n_in,
                              void* d_out, int out_size, void* d_ws, size_t ws_size,
                              hipStream_t stream) {
    const float* qs    = (const float*)d_in[0];
    const float* ks    = (const float*)d_in[1];
    const float* vs    = (const float*)d_in[2];
    const float* vnode = (const float*)d_in[3];
    const float* qs_s  = (const float*)d_in[4];
    const float* ks_s  = (const float*)d_in[5];
    const float* qs_t  = (const float*)d_in[6];
    const float* ks_t  = (const float*)d_in[7];
    const float* Ws1   = (const float*)d_in[8];
    const float* bs1   = (const float*)d_in[9];
    const float* Ws2   = (const float*)d_in[10];
    const float* bs2   = (const float*)d_in[11];
    const float* Ws3   = (const float*)d_in[12];
    const float* bs3   = (const float*)d_in[13];
    const float* Wv    = (const float*)d_in[14];
    const float* bv    = (const float*)d_in[15];
    const int* valid_lens = (const int*)d_in[16];

    // Scratch layout (floats): a_q | a_k | vproj | scores  (~8.5 MB total)
    float* ws     = (float*)d_ws;
    float* a_q    = ws;
    float* a_k    = a_q   + (size_t)B_ * Q_ * H1_;
    float* vproj  = a_k   + (size_t)B_ * K_ * H1_;
    float* scores = vproj + (size_t)B_ * K_ * H_;

    float* ctx  = (float*)d_out;                       // (B,Q,H)
    float* attn = ctx + (size_t)B_ * Q_ * H_;          // (B,Q,K)

    precompute_aq_kernel<<<(B_ * Q_ * H1_) / 256, 256, 0, stream>>>(qs, vnode, Ws1, bs1, a_q);
    precompute_ak_kernel<<<(B_ * K_ * H1_) / 256, 256, 0, stream>>>(ks, Ws1, a_k);
    precompute_vproj_kernel<<<(B_ * K_ * H_) / 256, 256, 0, stream>>>(vs, Wv, bv, vproj);

    dim3 grid(K_ / 16, Q_ / 16, B_);
    stmlp_scores_kernel<<<grid, 256, 0, stream>>>(a_q, a_k, Ws1, Ws2, bs2, Ws3, bs3,
                                                  qs_s, ks_s, qs_t, ks_t, valid_lens, scores);

    softmax_ctx_kernel<<<B_ * Q_, 256, 0, stream>>>(scores, vproj, attn, ctx);
}
